// AlignmentLayer_28776280883561
// MI455X (gfx1250) — matrix operand order repaired and checked by reading, not compiled, not run on hardware
//
#include <hip/hip_runtime.h>

typedef __attribute__((ext_vector_type(16))) _Float16 v16h;
typedef __attribute__((ext_vector_type(8)))  _Float16 v8h;
typedef __attribute__((ext_vector_type(4)))  _Float16 v4h;
typedef __attribute__((ext_vector_type(8)))  float    v8f;

#define BB   4
#define CC   2048
#define PP   4096
#define HIDN 256

static __device__ __forceinline__ v16h cmb(v8h lo, v8h hi) {
  return __builtin_shufflevector(lo, hi, 0,1,2,3,4,5,6,7,8,9,10,11,12,13,14,15);
}
static __device__ __forceinline__ v8f wmma_f16(v16h a, v16h b, v8f c) {
  // D = A(16x32 f16) * B(32x16 f16) + C(16x16 f32)
  return __builtin_amdgcn_wmma_f32_16x16x32_f16(false, a, false, b, (short)0, c,
                                                false, false);
}

// ---------------------------------------------------------------- fp32 -> f16
__global__ __launch_bounds__(256) void cvt_kernel(const float* __restrict__ src,
                                                  _Float16* __restrict__ dst,
                                                  int n4) {
  int i = blockIdx.x * 256 + threadIdx.x;
  if (i >= n4) return;
  float4 v = ((const float4*)src)[i];
  v4h o;
  o.x = (_Float16)v.x; o.y = (_Float16)v.y;
  o.z = (_Float16)v.z; o.w = (_Float16)v.w;
  ((v4h*)dst)[i] = o;
}

// -------------------------------------------------- projection: Qt/Kt[b][p][o]
// out[b,p,o] = sum_c W[o,c] * X[b,c,p] + bias[o]    (M = p, N = o, K = c)
__global__ __launch_bounds__(128) void proj_kernel(const float* __restrict__ query,
                                                   const float* __restrict__ prompt,
                                                   const _Float16* __restrict__ Wh,
                                                   const float* __restrict__ bias,
                                                   _Float16* __restrict__ Qt,
                                                   _Float16* __restrict__ Kt) {
  __shared__ _Float16 xT[64 * 32];  // X tile transposed: [p_local][c_local]

  const int tid  = threadIdx.x;
  const int wave = tid >> 5;
  const int lane = tid & 31;
  const int pBase = blockIdx.x * 64;
  const int oBase = blockIdx.y * 64;
  const int z = blockIdx.z;
  const int b = z & 3;
  const bool isQ = (z < 4);

  const float*   Xb = (isQ ? query : prompt) + (size_t)b * CC * PP;
  _Float16*      Ob = (isQ ? Qt : Kt) + (size_t)b * PP * HIDN;

  const int m      = lane & 15;
  const int off0   = (lane < 16) ? 0 : 16;   // A fragment K-run start
  const int offB   = (lane < 16) ? 0 : 16;   // B fragment K-run start
  v8f acc[4] = {};

  for (int c0 = 0; c0 < CC; c0 += 32) {
    __syncthreads();
    // stage X tile (32c x 64p) -> LDS transposed [p][c], converting to f16
    for (int idx = tid; idx < 32 * 16; idx += 128) {
      int cr = idx >> 4;
      int j  = (idx & 15) * 4;
      float4 v = *(const float4*)(Xb + (size_t)(c0 + cr) * PP + pBase + j);
      xT[(j + 0) * 32 + cr] = (_Float16)v.x;
      xT[(j + 1) * 32 + cr] = (_Float16)v.y;
      xT[(j + 2) * 32 + cr] = (_Float16)v.z;
      xT[(j + 3) * 32 + cr] = (_Float16)v.w;
    }
    __syncthreads();

    // A fragment: 16 p-rows (this wave) x 32 c, from LDS
    const _Float16* ap = &xT[(wave * 16 + m) * 32];
    v8h alo = *(const v8h*)(ap + off0);
    v8h ahi = *(const v8h*)(ap + off0 + 8);
    v16h a  = cmb(alo, ahi);

    // B fragments: 32 c x 16 o, contiguous rows of Wh
#pragma unroll
    for (int nt = 0; nt < 4; nt++) {
      int o = oBase + nt * 16 + m;
      const _Float16* wp = Wh + (size_t)o * CC + c0 + offB;
      v8h blo = *(const v8h*)wp;
      v8h bhi = *(const v8h*)(wp + 8);
      acc[nt] = wmma_f16(a, cmb(blo, bhi), acc[nt]);
    }
  }

  // epilogue: + bias, store f16 position-major
  const int rowAdd = (lane < 16) ? 0 : 8;
#pragma unroll
  for (int nt = 0; nt < 4; nt++) {
    int o = oBase + nt * 16 + m;
    float bv = bias[o];
#pragma unroll
    for (int r = 0; r < 8; r++) {
      int p = pBase + wave * 16 + r + rowAdd;
      Ob[(size_t)p * HIDN + o] = (_Float16)(acc[nt][r] + bv);
    }
  }
}

// ------------------------------------------------------- fused attention
// One WG = 32 query rows of one batch. 8 waves, 256 threads.
// LDS: S[32][4096] f16 score/prob panel (256 KB) + reduction scratch.
__global__ __launch_bounds__(256) void attn_kernel(const _Float16* __restrict__ Qt,
                                                   const _Float16* __restrict__ Kt,
                                                   const _Float16* __restrict__ Vh,
                                                   float* __restrict__ Out) {
  extern __shared__ _Float16 smem[];
  _Float16* S        = smem;                       // 32*4096 halves (256 KB)
  float*    red      = (float*)(smem + 32 * PP);   // 32 rows x 8 segs
  float*    rowmax   = red + 256;                  // 32
  float*    rowsumI  = rowmax + 32;                // 32

  const int tid  = threadIdx.x;
  const int wave = tid >> 5;
  const int lane = tid & 31;
  const int b  = blockIdx.y;
  const int q0 = blockIdx.x * 32;

  const _Float16* Qb = Qt + ((size_t)b * PP + q0) * HIDN;
  const _Float16* Kb = Kt + (size_t)b * PP * HIDN;
  const _Float16* Vb = Vh + (size_t)b * CC * PP;
  float* OutB = Out + (size_t)b * CC * PP;

  const int m      = lane & 15;
  const int off0   = (lane < 16) ? 0 : 16;
  const int offB   = (lane < 16) ? 0 : 16;
  const int rowAdd = (lane < 16) ? 0 : 8;

  // ---- phase 1: S = Q^T K  (each wave: 512 key columns, 2 query M-tiles)
#pragma unroll
  for (int mt = 0; mt < 2; mt++) {
    v16h aq[8];
#pragma unroll
    for (int kc = 0; kc < 8; kc++) {
      const _Float16* qp = Qb + (size_t)(mt * 16 + m) * HIDN + kc * 32 + off0;
      aq[kc] = cmb(*(const v8h*)qp, *(const v8h*)(qp + 8));
    }
    for (int j = 0; j < 32; j++) {
      int pk0 = wave * 512 + j * 16;
      const _Float16* kp = Kb + (size_t)(pk0 + m) * HIDN + offB;
      v8f acc = {};
#pragma unroll
      for (int kc = 0; kc < 8; kc++) {
        v8h blo = *(const v8h*)(kp + kc * 32);
        v8h bhi = *(const v8h*)(kp + kc * 32 + 8);
        acc = wmma_f16(aq[kc], cmb(blo, bhi), acc);
      }
#pragma unroll
      for (int r = 0; r < 8; r++)
        S[(size_t)(mt * 16 + r + rowAdd) * PP + pk0 + m] = (_Float16)acc[r];
    }
  }
  __syncthreads();

  // ---- phase 2: softmax (exact, two-pass), probs back to LDS as f16
  {
    const int row = tid >> 3;        // 0..31
    const int seg = tid & 7;         // 512 elements per (row,seg)
    _Float16* srow = S + (size_t)row * PP + seg * 512;

    float mx = -1e30f;
    for (int i = 0; i < 512; i += 8) {
      v8h v = *(const v8h*)(srow + i);
#pragma unroll
      for (int e = 0; e < 8; e++) mx = fmaxf(mx, (float)v[e]);
    }
    red[row * 8 + seg] = mx;
    __syncthreads();
    if (tid < 32) {
      float m2 = red[tid * 8];
      for (int k = 1; k < 8; k++) m2 = fmaxf(m2, red[tid * 8 + k]);
      rowmax[tid] = m2;
    }
    __syncthreads();

    float rmax = rowmax[row];
    float sum = 0.f;
    for (int i = 0; i < 512; i += 8) {
      v8h v = *(const v8h*)(srow + i);
      v8h e;
#pragma unroll
      for (int k = 0; k < 8; k++) {
        float ev = __expf((float)v[k] - rmax);
        sum += ev;
        e[k] = (_Float16)ev;
      }
      *(v8h*)(srow + i) = e;   // unnormalized probs; 1/sum folded into epilogue
    }
    red[row * 8 + seg] = sum;
    __syncthreads();
    if (tid < 32) {
      float s = 0.f;
      for (int k = 0; k < 8; k++) s += red[tid * 8 + k];
      rowsumI[tid] = 1.0f / s;
    }
    __syncthreads();
  }

  // ---- phase 3: O = P @ V^T
  // Each wave: 256 channels = 16 N-tiles, processed in 4 groups of 4 so each
  // A-fragment pair (2 query M-tiles) feeds 8 WMMAs per K-step.
  float rinv[16];
#pragma unroll
  for (int mt = 0; mt < 2; mt++)
#pragma unroll
    for (int r = 0; r < 8; r++)
      rinv[mt * 8 + r] = rowsumI[mt * 16 + rowAdd + r];

  for (int g = 0; g < 4; g++) {
    const _Float16* vp[4];
#pragma unroll
    for (int i = 0; i < 4; i++) {
      int c = wave * 256 + (g * 4 + i) * 16 + m;
      vp[i] = Vb + (size_t)c * PP + offB;
      __builtin_prefetch(vp[i], 0, 1);   // global_prefetch_b8 on the V stream
    }
    v8f acc[8] = {};  // [mt][i]
    for (int p0 = 0; p0 < PP; p0 += 32) {
      const _Float16* ap0 = S + (size_t)m * PP + p0 + off0;
      const _Float16* ap1 = S + (size_t)(16 + m) * PP + p0 + off0;
      v16h a0 = cmb(*(const v8h*)ap0, *(const v8h*)(ap0 + 8));
      v16h a1 = cmb(*(const v8h*)ap1, *(const v8h*)(ap1 + 8));
#pragma unroll
      for (int i = 0; i < 4; i++) {
        v8h blo = *(const v8h*)(vp[i] + p0);
        v8h bhi = *(const v8h*)(vp[i] + p0 + 8);
        v16h bm = cmb(blo, bhi);
        acc[i]     = wmma_f16(a0, bm, acc[i]);
        acc[4 + i] = wmma_f16(a1, bm, acc[4 + i]);
      }
    }
    // vectorized epilogue: rows of a tile are consecutive q -> 2x b128 stores
#pragma unroll
    for (int mt = 0; mt < 2; mt++) {
#pragma unroll
      for (int i = 0; i < 4; i++) {
        int c = wave * 256 + (g * 4 + i) * 16 + m;
        float* op = OutB + (size_t)c * PP + q0 + mt * 16 + rowAdd;
        v8f t = acc[mt * 4 + i];
        float4 lo4, hi4;
        lo4.x = t[0] * rinv[mt * 8 + 0];
        lo4.y = t[1] * rinv[mt * 8 + 1];
        lo4.z = t[2] * rinv[mt * 8 + 2];
        lo4.w = t[3] * rinv[mt * 8 + 3];
        hi4.x = t[4] * rinv[mt * 8 + 4];
        hi4.y = t[5] * rinv[mt * 8 + 5];
        hi4.z = t[6] * rinv[mt * 8 + 6];
        hi4.w = t[7] * rinv[mt * 8 + 7];
        *(float4*)op       = lo4;
        *(float4*)(op + 4) = hi4;
      }
    }
  }
}

// ---------------------------------------------------------------- launcher
extern "C" void kernel_launch(void* const* d_in, const int* in_sizes, int n_in,
                              void* d_out, int out_size, void* d_ws, size_t ws_size,
                              hipStream_t stream) {
  const float* query  = (const float*)d_in[0];  // (4,2048,64,64)
  const float* prompt = (const float*)d_in[1];  // (4,2048,64,64)
  const float* Wf     = (const float*)d_in[2];  // (256,2048)
  const float* bias   = (const float*)d_in[3];  // (256,)
  float* out = (float*)d_out;                   // (4,2048,64,64) fp32

  // workspace layout (f16): Wh | Qt | Kt | Vh  (~81 MB)
  _Float16* Wh = (_Float16*)d_ws;
  _Float16* Qt = Wh + (size_t)HIDN * CC;
  _Float16* Kt = Qt + (size_t)BB * PP * HIDN;
  _Float16* Vh = Kt + (size_t)BB * PP * HIDN;

  // 1) convert W and prompt(V) to f16
  {
    int n4 = (HIDN * CC) / 4;
    cvt_kernel<<<(n4 + 255) / 256, 256, 0, stream>>>(Wf, Wh, n4);
  }
  {
    int n4 = (BB * CC * PP) / 4;
    cvt_kernel<<<(n4 + 255) / 256, 256, 0, stream>>>(prompt, Vh, n4);
  }

  // 2) projection -> Qt, Kt  (position-major f16)
  proj_kernel<<<dim3(PP / 64, HIDN / 64, BB * 2), 128, 0, stream>>>(
      query, prompt, Wh, bias, Qt, Kt);

  // 3) fused attention: 32-query score panel in LDS (256 KB), WMMA both GEMMs
  size_t smem = (size_t)32 * PP * sizeof(_Float16) + (256 + 32 + 32) * sizeof(float);
  attn_kernel<<<dim3(PP / 32, BB), 256, smem, stream>>>(Qt, Kt, Vh, out);
}
